// WDHA_18098992185358
// MI455X (gfx1250) — compile-verified, matrix operand order checked by reading, not run on hardware
//
#include <hip/hip_runtime.h>
#include <hip/hip_bf16.h>
#include <cstddef>
#include <cstdint>

typedef __attribute__((ext_vector_type(16))) _Float16 v16h;
typedef __attribute__((ext_vector_type(8)))  _Float16 v8h;
typedef __attribute__((ext_vector_type(8)))  float    v8f;

constexpr int B_ = 32, F_ = 64, T_ = 518, H_ = 64, G_ = 256, C_ = 128;
constexpr int NW_ = 6, WIDE_ = 15, PADW_ = 7;
constexpr int WC_ = NW_ * C_;            // 768
constexpr int L2_ = 516, L3_ = 512, L4_ = 510;

// ISA 7.12.2, 16-bit A/B 16x32 fragment: lane kg = lane>>4; halves j=0..7 hold
// K = kb0..kb0+7 (kb0 = kk + kg*8), halves j=8..15 hold K = kb1..kb1+7
// (kb1 = kk + 16 + kg*8). So a fragment is two contiguous 8-half (16B) runs.
__device__ __forceinline__ int frag_koff(int j, int kg) {
  int v = j >> 1, p = j & 1;
  return ((v < 4) ? 0 : 16) + kg * 8 + 2 * (v & 3) + p;
}
__device__ __forceinline__ v16h frag_cat(v8h lo, v8h hi) {
  v16h r;
#pragma unroll
  for (int j = 0; j < 8; ++j) { r[j] = lo[j]; r[j + 8] = hi[j]; }
  return r;
}
// LDS byte offset of a generic (flat) shared pointer: flat->LDS keeps addr[31:0].
__device__ __forceinline__ unsigned lds_off(const void* p) {
  return (unsigned)(uintptr_t)p;
}

// ---------------------------------------------------------------------------
// f16 packing helpers (weights once per call; x transposed [B,F,T]->[B,T,F])
// ---------------------------------------------------------------------------
__global__ void k_cvt_f16(const float* __restrict__ in, _Float16* __restrict__ out, int n) {
  int i = blockIdx.x * blockDim.x + threadIdx.x;
  if (i < n) out[i] = (_Float16)in[i];
}
__global__ void k_pack_xT(const float* __restrict__ x, _Float16* __restrict__ xh) {
  int i = blockIdx.x * blockDim.x + threadIdx.x;   // output index (b*T+t)*F+f
  if (i >= B_ * T_ * F_) return;
  int f = i % F_;
  int t = (i / F_) % T_;
  int b = i / (F_ * T_);
  xh[i] = (_Float16)x[((size_t)b * F_ + f) * T_ + t];
}

// ---------------------------------------------------------------------------
// K1: LSTM input projection, xg = x_dir @ wih^T + bias.
//     M=B*T=16576, N=G=256, K=F=64. A/B f16 pre-packed -> b128 fragment loads.
// ---------------------------------------------------------------------------
__global__ void k_lstm_proj(const _Float16* __restrict__ xh,
                            const _Float16* __restrict__ wifh, const float* __restrict__ bf,
                            const _Float16* __restrict__ wirh, const float* __restrict__ br,
                            float* __restrict__ xg_f, float* __restrict__ xg_r) {
  const int wave = (int)((blockIdx.x * blockDim.x + threadIdx.x) >> 5);
  const int lane = threadIdx.x & 31;
  const int mtiles = (B_ * T_) / 16;       // 1036
  const int ntiles = G_ / 16;              // 16
  const int per_dir = mtiles * ntiles;
  if (wave >= 2 * per_dir) return;
  const int dir = wave / per_dir;
  const int rem = wave % per_dir;
  const int mt = rem / ntiles, nt = rem % ntiles;

  const _Float16* wih  = dir ? wirh : wifh;
  const float*    bias = dir ? br   : bf;
  float*          xg   = dir ? xg_r : xg_f;

  const int kg   = lane >> 4;
  const int mrow = mt * 16 + (lane & 15);
  const int ab   = mrow / T_;
  const int as   = mrow % T_;
  const int at   = dir ? (T_ - 1 - as) : as;        // reversed input for bwd dir
  const int ncol = nt * 16 + (lane & 15);

  const _Float16* arow = xh + ((size_t)ab * T_ + at) * F_;
  const _Float16* brow = wih + (size_t)ncol * F_;

  v8f acc = {};
#pragma unroll
  for (int kk = 0; kk < F_; kk += 32) {
    v16h af = frag_cat(*(const v8h*)(arow + kk + kg * 8),
                       *(const v8h*)(arow + kk + 16 + kg * 8));
    v16h bfr = frag_cat(*(const v8h*)(brow + kk + kg * 8),
                        *(const v8h*)(brow + kk + 16 + kg * 8));
    acc = __builtin_amdgcn_wmma_f32_16x16x32_f16(false, af, false, bfr,
                                                 (short)0, acc, false, false);
  }
#pragma unroll
  for (int i = 0; i < 8; ++i) {
    int row = mt * 16 + i + ((lane >= 16) ? 8 : 0);
    int g   = nt * 16 + (lane & 15);
    int b   = row / T_, s = row % T_;
    xg[((size_t)b * T_ + s) * G_ + g] = acc[i] + bias[g];
  }
}

// ---------------------------------------------------------------------------
// K2: LSTM recurrence. 2 persistent workgroups; h (f16), c, gates in LDS.
//     Per step: gates = xg[:,s,:] + h @ whh^T via WMMA (A: ds_load_b128).
// ---------------------------------------------------------------------------
__global__ void k_lstm_rec(const float* __restrict__ xg_f, const float* __restrict__ xg_r,
                           const _Float16* __restrict__ whfh, const _Float16* __restrict__ whrh,
                           float* __restrict__ X) {
  const int dir = blockIdx.x;
  const float*    xg  = dir ? xg_r : xg_f;
  const _Float16* whh = dir ? whrh : whfh;

  __shared__ _Float16 hsh[B_][H_];     // 4 KB, b128-aligned rows (128B)
  __shared__ float    cs[B_][H_];      // 8 KB
  __shared__ float    gs[B_][G_];      // 32 KB

  const int tid = threadIdx.x;
  for (int i = tid; i < B_ * H_; i += blockDim.x) {
    hsh[i >> 6][i & 63] = (_Float16)0.f;
    cs[i >> 6][i & 63] = 0.f;
  }
  __syncthreads();

  const int wave = tid >> 5, lane = tid & 31;
  const int kg = lane >> 4;

  for (int s = 0; s < T_; ++s) {
    for (int q = 0; q < 4; ++q) {                    // 32 tiles, 4 per wave
      int tile = wave * 4 + q;
      int mt = tile >> 4, nt = tile & 15;
      int ncol = nt * 16 + (lane & 15);
      const _Float16* arow = &hsh[mt * 16 + (lane & 15)][0];
      const _Float16* brow = whh + (size_t)ncol * H_;
      v8f acc;
#pragma unroll
      for (int i = 0; i < 8; ++i) {                  // C init = precomputed xg
        int row = mt * 16 + i + ((lane >= 16) ? 8 : 0);
        acc[i] = xg[((size_t)row * T_ + s) * G_ + ncol];
      }
#pragma unroll
      for (int kk = 0; kk < H_; kk += 32) {
        v16h af = frag_cat(*(const v8h*)(arow + kk + kg * 8),
                           *(const v8h*)(arow + kk + 16 + kg * 8));
        v16h bfr = frag_cat(*(const v8h*)(brow + kk + kg * 8),
                            *(const v8h*)(brow + kk + 16 + kg * 8));
        acc = __builtin_amdgcn_wmma_f32_16x16x32_f16(false, af, false, bfr,
                                                     (short)0, acc, false, false);
      }
#pragma unroll
      for (int i = 0; i < 8; ++i) {
        int row = mt * 16 + i + ((lane >= 16) ? 8 : 0);
        gs[row][ncol] = acc[i];
      }
    }
    __syncthreads();

    const int t_out = dir ? (T_ - 1 - s) : s;
    for (int e = tid; e < B_ * H_; e += blockDim.x) {
      int b = e >> 6, j = e & 63;
      float gi = gs[b][j], gf = gs[b][H_ + j], gg = gs[b][2 * H_ + j], go = gs[b][3 * H_ + j];
      float si = 1.f / (1.f + __expf(-gi));
      float sf = 1.f / (1.f + __expf(-gf));
      float so = 1.f / (1.f + __expf(-go));
      float cc = sf * cs[b][j] + si * tanhf(gg);
      float hh = so * tanhf(cc);
      cs[b][j] = cc;
      hsh[b][j] = (_Float16)hh;
      X[(size_t)b * C_ * T_ + (size_t)(dir * H_ + j) * T_ + t_out] = hh;
    }
    __syncthreads();
  }
}

// ---------------------------------------------------------------------------
// K3: wave layer — dynamic cosine conv. One thread per (b,c,t).
// ---------------------------------------------------------------------------
__global__ void k_wave(const float* __restrict__ X, const float* __restrict__ base_w,
                       const float* __restrict__ base_b, float* __restrict__ wv) {
  int idx = blockIdx.x * blockDim.x + threadIdx.x;
  if (idx >= B_ * C_ * T_) return;
  int t = idx % T_;
  int c = (idx / T_) % C_;
  int b = idx / (T_ * C_);
  const float* xr = X + (size_t)b * C_ * T_ + (size_t)c * T_;

  float win[WIDE_];
#pragma unroll
  for (int k = 0; k < WIDE_; ++k) {
    int tt = t - PADW_ + k;
    win[k] = (tt >= 0 && tt < T_) ? xr[tt] : 0.f;
  }
#pragma unroll
  for (int ii = 0; ii < NW_; ++ii) {
    float sc = base_b[ii];
#pragma unroll
    for (int k = 0; k < WIDE_; ++k) sc += win[k] * base_w[ii * WIDE_ + k];
    sc = fmaxf(sc, 0.f);
    float tpf = 6.28318530717958647f * (float)(ii + 1) * sc;  // FREQS = 1..6
    float o = 0.f;
#pragma unroll
    for (int k = 0; k < WIDE_; ++k) {
      float off = (float)(k - PADW_) * (1.0f / WIDE_);
      o += win[k] * __cosf(tpf * off);
    }
    float sq = sqrtf((truncf(sc * WIDE_) + 1.f) * (2.f / (WIDE_ + 2)));
    wv[(size_t)b * WC_ * T_ + (size_t)(ii * C_ + c) * T_ + t] = o / sq;
  }
}

// ---------------------------------------------------------------------------
// BatchNorm (training-mode): per-channel stats over (B, L), then in-place apply.
// ---------------------------------------------------------------------------
__global__ void k_bn_stats(const float* __restrict__ x, int C, int L,
                           float* __restrict__ mean, float* __restrict__ istd) {
  int c = blockIdx.x;
  float s = 0.f, s2 = 0.f;
  for (int i = threadIdx.x; i < B_ * L; i += blockDim.x) {
    int b = i / L, l = i % L;
    float v = x[(size_t)b * C * L + (size_t)c * L + l];
    s += v; s2 += v * v;
  }
  __shared__ float rs[256], rs2[256];
  rs[threadIdx.x] = s; rs2[threadIdx.x] = s2;
  __syncthreads();
  for (int w = 128; w > 0; w >>= 1) {
    if (threadIdx.x < w) { rs[threadIdx.x] += rs[threadIdx.x + w]; rs2[threadIdx.x] += rs2[threadIdx.x + w]; }
    __syncthreads();
  }
  if (threadIdx.x == 0) {
    float n = (float)(B_ * L);
    float m = rs[0] / n;
    float v = rs2[0] / n - m * m;
    mean[c] = m;
    istd[c] = rsqrtf(fmaxf(v, 0.f) + 1e-5f);
  }
}

__global__ void k_bn_apply(float* __restrict__ x, int C, int L,
                           const float* __restrict__ mean, const float* __restrict__ istd,
                           const float* __restrict__ g, const float* __restrict__ be) {
  size_t idx = (size_t)blockIdx.x * blockDim.x + threadIdx.x;
  size_t n = (size_t)B_ * (size_t)C * (size_t)L;
  if (idx >= n) return;
  int c = (int)((idx / (size_t)L) % (size_t)C);
  x[idx] = (x[idx] - mean[c]) * istd[c] * g[c] + be[c];
}

// ---------------------------------------------------------------------------
// Conv1d(VALID)+ReLU as implicit GEMM. Block covers all M-tiles x ntpb N-tiles
// so the im2col B-chunk is staged in LDS ONCE per block via the CDNA5 async
// memory->LDS path (global_load_async_to_lds_b32, ASYNCcnt), then each wave
// feeds WMMA with b128 A fragments (f16 weights) and LDS B reads.
// ---------------------------------------------------------------------------
__global__ void k_conv_wmma(const float* __restrict__ in, const _Float16* __restrict__ w_h,
                            const float* __restrict__ bias, float* __restrict__ out,
                            int Cin, int Lin, int Cout, int Lout, int KW,
                            int ntpb, int kchunk) {
  extern __shared__ float ldsB[];                  // [kchunk][ncols] f32
  const int K = Cin * KW;
  const int ncols = ntpb * 16;
  const int tid = threadIdx.x;
  const int wave = tid >> 5, lane = tid & 31;
  const int ntq = wave % ntpb, mt = wave / ntpb;
  const int n0 = blockIdx.x * ncols;
  const int kg = lane >> 4;
  const int arow = mt * 16 + (lane & 15);          // output channel (A row)
  const int colB = ntq * 16 + (lane & 15);         // column within block tile
  const unsigned lds_base = lds_off(&ldsB[0]);

  v8f acc = {};
  for (int k0 = 0; k0 < K; k0 += kchunk) {
    // ---- stage B chunk: async global->LDS, 4B-aligned f32 elements ----
    const int elems = kchunk * ncols;              // multiple of blockDim
    for (int idx = tid; idx < elems; idx += blockDim.x) {
      int krel = idx / ncols;
      int col  = idx - krel * ncols;
      int n  = n0 + col;
      int bb = n / Lout, ll = n - bb * Lout;
      int k  = k0 + krel;
      int ic = k / KW, kw = k - ic * KW;
      const float* src = in + ((size_t)bb * Cin + ic) * Lin + ll + kw;
      unsigned dst = lds_base + (unsigned)(idx * 4);
      asm volatile("global_load_async_to_lds_b32 %0, %1, off"
                   :: "v"(dst), "v"(src) : "memory");
    }
    asm volatile("s_wait_asynccnt 0x0" ::: "memory");
    __syncthreads();
    // ---- compute: kchunk/32 WMMA steps per wave-tile ----
    const _Float16* aw = w_h + (size_t)arow * K + k0;
    for (int krel = 0; krel < kchunk; krel += 32) {
      v16h af = frag_cat(*(const v8h*)(aw + krel + kg * 8),
                         *(const v8h*)(aw + krel + 16 + kg * 8));
      v16h bfr;
#pragma unroll
      for (int j = 0; j < 16; ++j)
        bfr[j] = (_Float16)ldsB[(krel + frag_koff(j, kg)) * ncols + colB];
      acc = __builtin_amdgcn_wmma_f32_16x16x32_f16(false, af, false, bfr,
                                                   (short)0, acc, false, false);
    }
    __syncthreads();                               // LDS reuse next chunk
  }
#pragma unroll
  for (int i = 0; i < 8; ++i) {
    int oc = mt * 16 + i + ((lane >= 16) ? 8 : 0);
    int n  = n0 + ntq * 16 + (lane & 15);
    int b  = n / Lout, l = n - b * Lout;
    out[(size_t)b * Cout * Lout + (size_t)oc * Lout + l] = fmaxf(acc[i] + bias[oc], 0.f);
  }
}

// ---------------------------------------------------------------------------
// DWT branch: Haar commutes with the channel mean, so reduce channels first.
// ---------------------------------------------------------------------------
__global__ void k_dwt(const float* __restrict__ ob, float* __restrict__ dwt) {
  const int b = blockIdx.x;
  __shared__ float m[L3_];
  __shared__ float a1[256], d1[256], a2[128], d2[128], a3[64], d3[64], a4[32], d4[32];
  const float s = 0.70710678118654752f;
  for (int l = threadIdx.x; l < L3_; l += blockDim.x) {
    float acc = 0.f;
    for (int c = 0; c < 32; ++c) acc += ob[(size_t)b * 32 * L3_ + (size_t)c * L3_ + l];
    m[l] = acc * (1.f / 32.f);
  }
  __syncthreads();
  for (int l = threadIdx.x; l < 256; l += blockDim.x) { a1[l] = (m[2*l] + m[2*l+1]) * s;  d1[l] = (m[2*l] - m[2*l+1]) * s; }
  __syncthreads();
  for (int l = threadIdx.x; l < 128; l += blockDim.x) { a2[l] = (a1[2*l] + a1[2*l+1]) * s; d2[l] = (a1[2*l] - a1[2*l+1]) * s; }
  __syncthreads();
  for (int l = threadIdx.x; l < 64;  l += blockDim.x) { a3[l] = (a2[2*l] + a2[2*l+1]) * s; d3[l] = (a2[2*l] - a2[2*l+1]) * s; }
  __syncthreads();
  for (int l = threadIdx.x; l < 32;  l += blockDim.x) { a4[l] = (a3[2*l] + a3[2*l+1]) * s; d4[l] = (a3[2*l] - a3[2*l+1]) * s; }
  __syncthreads();
  float* dst = dwt + (size_t)b * 5 * L3_;
  for (int l = threadIdx.x; l < L3_; l += blockDim.x) {
    dst[0 * L3_ + l] = a4[l >> 4];
    dst[1 * L3_ + l] = d4[l >> 4];
    dst[2 * L3_ + l] = d3[l >> 3];
    dst[3 * L3_ + l] = d2[l >> 2];
    dst[4 * L3_ + l] = d1[l >> 1];
  }
}

__global__ void k_conv5(const float* __restrict__ dwt, const float* __restrict__ w,
                        const float* __restrict__ bias, float* __restrict__ out) {
  int idx = blockIdx.x * blockDim.x + threadIdx.x;
  if (idx >= B_ * 16 * L4_) return;
  int l  = idx % L4_;
  int oc = (idx / L4_) % 16;
  int b  = idx / (L4_ * 16);
  float acc = bias[oc];
#pragma unroll
  for (int ic = 0; ic < 5; ++ic)
#pragma unroll
    for (int kw = 0; kw < 3; ++kw)
      acc += w[oc * 15 + ic * 3 + kw] * dwt[(size_t)b * 5 * L3_ + (size_t)ic * L3_ + l + kw];
  out[idx] = fmaxf(acc, 0.f);
}

__global__ void k_pool(const float* __restrict__ c4, const float* __restrict__ c5,
                       float* __restrict__ pooled) {
  int bc = blockIdx.x;
  int b = bc >> 4, ch = bc & 15;
  const float* p4 = c4 + (size_t)b * 16 * L4_ + (size_t)ch * L4_;
  const float* p5 = c5 + (size_t)b * 16 * L4_ + (size_t)ch * L4_;
  float s = 0.f;
  for (int l = threadIdx.x; l < L4_; l += blockDim.x) s += p4[l] + p5[l];
  __shared__ float red[256];
  red[threadIdx.x] = s;
  __syncthreads();
  for (int w = 128; w > 0; w >>= 1) {
    if (threadIdx.x < w) red[threadIdx.x] += red[threadIdx.x + w];
    __syncthreads();
  }
  if (threadIdx.x == 0) pooled[bc] = red[0] / (float)(2 * L4_);
}

__global__ void k_fc(const float* __restrict__ pooled, const float* __restrict__ fcw,
                     const float* __restrict__ fcb, float* __restrict__ out) {
  int idx = blockIdx.x * blockDim.x + threadIdx.x;
  if (idx >= B_ * 10) return;
  int b = idx / 10, o = idx % 10;
  float acc = fcb[o];
#pragma unroll
  for (int ch = 0; ch < 16; ++ch) acc += pooled[b * 16 + ch] * fcw[o * 16 + ch];
  out[idx] = acc;
}

// ---------------------------------------------------------------------------
extern "C" void kernel_launch(void* const* d_in, const int* in_sizes, int n_in,
                              void* d_out, int out_size, void* d_ws, size_t ws_size,
                              hipStream_t stream) {
  const float* x    = (const float*)d_in[0];
  const float* wif  = (const float*)d_in[1];
  const float* whf  = (const float*)d_in[2];
  const float* bf   = (const float*)d_in[3];
  const float* wir  = (const float*)d_in[4];
  const float* whr  = (const float*)d_in[5];
  const float* br   = (const float*)d_in[6];
  const float* basw = (const float*)d_in[7];
  const float* basb = (const float*)d_in[8];
  const float* g1 = (const float*)d_in[9],  *be1 = (const float*)d_in[10];
  const float* g2 = (const float*)d_in[11], *be2 = (const float*)d_in[12];
  const float* g3 = (const float*)d_in[13], *be3 = (const float*)d_in[14];
  const float* g4 = (const float*)d_in[15], *be4 = (const float*)d_in[16];
  const float* g5 = (const float*)d_in[17], *be5 = (const float*)d_in[18];
  const float* w2 = (const float*)d_in[19], *b2 = (const float*)d_in[20];
  const float* w3 = (const float*)d_in[21], *b3 = (const float*)d_in[22];
  const float* w4 = (const float*)d_in[23], *b4 = (const float*)d_in[24];
  const float* w5 = (const float*)d_in[25], *b5 = (const float*)d_in[26];
  const float* fcw = (const float*)d_in[27], *fcb = (const float*)d_in[28];
  float* out = (float*)d_out;

  // ---- f32 workspace (wave aliases the dead xg region) ----
  float* ws   = (float*)d_ws;
  float* X    = ws;                             // [B,C,T]
  float* xgf  = X    + (size_t)B_ * C_ * T_;    // [B,T,G]
  float* xgr  = xgf  + (size_t)B_ * T_ * G_;
  float* wv   = xgf;                            // [B,768,T] (aliases xg)
  float* c2   = xgf  + (size_t)B_ * WC_ * T_;   // [B,128,516]
  float* c3   = c2   + (size_t)B_ * 128 * L2_;  // [B,32,512]
  float* c4   = c3   + (size_t)B_ * 32 * L3_;   // [B,16,510]
  float* dwtb = c4   + (size_t)B_ * 16 * L4_;   // [B,5,512]
  float* c5   = dwtb + (size_t)B_ * 5 * L3_;    // [B,16,510]
  float* pool = c5   + (size_t)B_ * 16 * L4_;   // 512
  float* mean = pool + 512;                     // 768
  float* istd = mean + WC_;                     // 768
  // ---- f16 packed region (16B aligned; all sizes multiples of 8 halves) ----
  uintptr_t hp = ((uintptr_t)(istd + WC_) + 15) & ~(uintptr_t)15;
  _Float16* xh   = (_Float16*)hp;               // [B,T,F] = 1,060,864
  _Float16* wifh = xh   + (size_t)B_ * T_ * F_; // 16384
  _Float16* wirh = wifh + G_ * F_;
  _Float16* whfh = wirh + G_ * F_;
  _Float16* whrh = whfh + G_ * H_;
  _Float16* w2h  = whrh + G_ * H_;              // 128*2304
  _Float16* w3h  = w2h  + 128 * WC_ * 3;        // 32*640
  _Float16* w4h  = w3h  + 32 * 128 * 5;         // 16*96

  // 0) pack weights + transposed input to f16 (WMMA-native operand precision)
  k_cvt_f16<<<(G_ * F_ + 255) / 256, 256, 0, stream>>>(wif, wifh, G_ * F_);
  k_cvt_f16<<<(G_ * F_ + 255) / 256, 256, 0, stream>>>(wir, wirh, G_ * F_);
  k_cvt_f16<<<(G_ * H_ + 255) / 256, 256, 0, stream>>>(whf, whfh, G_ * H_);
  k_cvt_f16<<<(G_ * H_ + 255) / 256, 256, 0, stream>>>(whr, whrh, G_ * H_);
  k_cvt_f16<<<(128 * WC_ * 3 + 255) / 256, 256, 0, stream>>>(w2, w2h, 128 * WC_ * 3);
  k_cvt_f16<<<(32 * 128 * 5 + 255) / 256, 256, 0, stream>>>(w3, w3h, 32 * 128 * 5);
  k_cvt_f16<<<(16 * 32 * 3 + 255) / 256, 256, 0, stream>>>(w4, w4h, 16 * 32 * 3);
  k_pack_xT<<<(B_ * T_ * F_ + 255) / 256, 256, 0, stream>>>(x, xh);

  // 1) LSTM input projections (WMMA, b128 f16 fragments)
  {
    int waves = 2 * ((B_ * T_) / 16) * (G_ / 16);   // 33152
    k_lstm_proj<<<(waves + 7) / 8, 256, 0, stream>>>(xh, wifh, bf, wirh, br, xgf, xgr);
  }
  // 2) Recurrence (persistent, WMMA gate GEMM/step, h f16 in LDS)
  k_lstm_rec<<<2, 256, 0, stream>>>(xgf, xgr, whfh, whrh, X);
  // 3) Wave layer (cos-heavy VALU)
  k_wave<<<(B_ * C_ * T_ + 255) / 256, 256, 0, stream>>>(X, basw, basb, wv);
  // 4) BN1
  k_bn_stats<<<WC_, 256, 0, stream>>>(wv, WC_, T_, mean, istd);
  k_bn_apply<<<(int)(((size_t)B_ * WC_ * T_ + 255) / 256), 256, 0, stream>>>(wv, WC_, T_, mean, istd, g1, be1);
  // 5) conv2 (768->128,k=3): 1032 blocks x (8mt x 1nt), K-chunk 576 (36KB LDS)
  k_conv_wmma<<<(B_ * L2_) / 16, 256, 576 * 16 * 4, stream>>>(wv, w2h, b2, c2, WC_, T_, 128, L2_, 3, 1, 576);
  k_bn_stats<<<128, 256, 0, stream>>>(c2, 128, L2_, mean, istd);
  k_bn_apply<<<(int)(((size_t)B_ * 128 * L2_ + 255) / 256), 256, 0, stream>>>(c2, 128, L2_, mean, istd, g2, be2);
  // 6) conv3 (128->32,k=5): 256 blocks x (2mt x 4nt), K-chunk 160 (40KB LDS)
  k_conv_wmma<<<(B_ * L3_) / 64, 256, 160 * 64 * 4, stream>>>(c2, w3h, b3, c3, 128, L2_, 32, L3_, 5, 4, 160);
  k_bn_stats<<<32, 256, 0, stream>>>(c3, 32, L3_, mean, istd);
  k_bn_apply<<<(int)(((size_t)B_ * 32 * L3_ + 255) / 256), 256, 0, stream>>>(c3, 32, L3_, mean, istd, g3, be3);
  // 7) conv4 (32->16,k=3): 255 blocks x (1mt x 4nt) @128 thr, K=96 (24KB LDS)
  k_conv_wmma<<<(B_ * L4_) / 64, 128, 96 * 64 * 4, stream>>>(c3, w4h, b4, c4, 32, L3_, 16, L4_, 3, 4, 96);
  k_bn_stats<<<16, 256, 0, stream>>>(c4, 16, L4_, mean, istd);
  k_bn_apply<<<(int)(((size_t)B_ * 16 * L4_ + 255) / 256), 256, 0, stream>>>(c4, 16, L4_, mean, istd, g4, be4);
  // 8) DWT branch + conv5 + BN5
  k_dwt<<<B_, 256, 0, stream>>>(c3, dwtb);
  k_conv5<<<(B_ * 16 * L4_ + 255) / 256, 256, 0, stream>>>(dwtb, w5, b5, c5);
  k_bn_stats<<<16, 256, 0, stream>>>(c5, 16, L4_, mean, istd);
  k_bn_apply<<<(int)(((size_t)B_ * 16 * L4_ + 255) / 256), 256, 0, stream>>>(c5, 16, L4_, mean, istd, g5, be5);
  // 9) pool + FC
  k_pool<<<B_ * 16, 256, 0, stream>>>(c4, c5, pool);
  k_fc<<<(B_ * 10 + 255) / 256, 256, 0, stream>>>(pool, fcw, fcb, out);
}
